// NeuralPredictor_58918361367209
// MI455X (gfx1250) — matrix operand order repaired and checked by reading, not compiled, not run on hardware
//
#include <hip/hip_runtime.h>
#include <math.h>

// ---------------- types ----------------
typedef _Float16 half16 __attribute__((ext_vector_type(16)));
typedef float    float8 __attribute__((ext_vector_type(8)));
typedef unsigned uint4v __attribute__((ext_vector_type(4)));

#define N_VARS 1500
#define N_FUNCS 6000
#define N_EDGES 18000
#define KP 160            // padded K for all layers (133/150 -> 160); 5 k-tiles of 32
#define KTILES 5
#define NP_H 160          // padded N for 150-wide layers (10 n-tiles)
#define NP_L4 128         // padded N for the 100-wide layer (8 n-tiles)
#define MEMH 150
#define DEC 100

// ---------------- kernels ----------------

// One-hot mask -> index. mode 0: idx[col]=row (per-column one-hot),
// mode 1: idx[row]=col (per-row one-hot). cols must be multiple of 4.
// Non-temporal loads: this is a 540MB use-once stream; keep it out of L2.
__global__ void np_extract_onehot(const float* __restrict__ mask, int* __restrict__ idx,
                                  int rows, int cols, int mode) {
    long long t = (long long)blockIdx.x * blockDim.x + threadIdx.x;
    long long total = ((long long)rows * cols) >> 2;
    if (t >= total) return;
    const uint4v* p = (const uint4v*)mask;
    uint4v v = __builtin_nontemporal_load(&p[t]);   // float bits; one-hot => 0x0 or 0x3f800000
    long long base = t * 4;
    int r = (int)(base / cols);
    int c = (int)(base % cols);
    if (v.x != 0u) { if (mode == 0) idx[c]     = r; else idx[r] = c;     }
    if (v.y != 0u) { if (mode == 0) idx[c + 1] = r; else idx[r] = c + 1; }
    if (v.z != 0u) { if (mode == 0) idx[c + 2] = r; else idx[r] = c + 2; }
    if (v.w != 0u) { if (mode == 0) idx[c + 3] = r; else idx[r] = c + 3; }
}

// Build f16 inputs Av/Af [E, KP]: [dec_state(100) | edge_feat(1) | meta(32) | zeros]
__global__ void np_pack_inputs(const float* __restrict__ dvs, const float* __restrict__ dfs,
                               const float* __restrict__ ef, const float* __restrict__ meta,
                               const int* __restrict__ var_of_edge, const int* __restrict__ g_of_var,
                               _Float16* __restrict__ Av, _Float16* __restrict__ Af) {
    int t = blockIdx.x * blockDim.x + threadIdx.x;  // E * (KP/2) threads, 2 cols each
    if (t >= N_EDGES * (KP / 2)) return;
    int e = t / (KP / 2), d = t % (KP / 2);
    int g = g_of_var[var_of_edge[e]];
    #pragma unroll
    for (int j = 0; j < 2; ++j) {
        int c = 2 * d + j;
        float fv, ff;
        if (c < DEC)            { fv = dvs[(size_t)e * DEC + c]; ff = dfs[(size_t)e * DEC + c]; }
        else if (c == DEC)      { fv = ff = ef[e]; }
        else if (c < DEC + 33)  { fv = ff = meta[g * 32 + (c - (DEC + 1))]; }
        else                    { fv = ff = 0.f; }
        Av[(size_t)e * KP + c] = (_Float16)fv;
        Af[(size_t)e * KP + c] = (_Float16)ff;
    }
}

// Pack f32 weight [K,N] row-major -> f16 B-fragment layout for v_wmma 16x16x32.
// Fragment dword index: (((kt*Ntiles)+nt)*32 + lane)*8 + i, where lane = 16*(k>=16 within tile)+n,
// VGPR i holds k-pair (16h + 2i, 16h + 2i + 1) of k-tile kt.
__global__ void np_pack_w(const float* __restrict__ W, unsigned* __restrict__ out,
                          int K, int N, int Ktiles, int Ntiles) {
    int t = blockIdx.x * blockDim.x + threadIdx.x;
    int total = Ktiles * Ntiles * 256;
    if (t >= total) return;
    int i    = t & 7;
    int lane = (t >> 3) & 31;
    int tile = t >> 8;
    int nt = tile % Ntiles;
    int kt = tile / Ntiles;
    int h = lane >> 4, nl = lane & 15;
    int n  = nt * 16 + nl;
    int k0 = kt * 32 + 16 * h + 2 * i;
    float f0 = (k0     < K && n < N) ? W[(size_t)k0 * N + n]       : 0.f;
    float f1 = (k0 + 1 < K && n < N) ? W[(size_t)(k0 + 1) * N + n] : 0.f;
    union { _Float16 h2[2]; unsigned u; } p;
    p.h2[0] = (_Float16)f0;
    p.h2[1] = (_Float16)f1;
    out[t] = p.u;
}

// One wave computes a 16 x (16*NG) strip: Out = act(A @ Wpack + bias).
// NG, NTILES, NP are compile-time so NTILES % NG == 0 (no remainder group, no
// conditional accumulators) and all fragment addresses fold to immediates.
// The A fragment is loaded once per k-tile and reused across NG wmma ops.
template <int NG, int NTILES, int NP>
__global__ __launch_bounds__(32)
void np_wmma_gemm(const _Float16* __restrict__ A, const unsigned* __restrict__ Wpack,
                  const float* __restrict__ bias, _Float16* __restrict__ Out,
                  int Mvalid, int Nvalid, int act) {
    const int nt0 = blockIdx.x * NG;
    const int mt  = blockIdx.y;
    const int lane = threadIdx.x;
    const int h = lane >> 4, nl = lane & 15;
    const unsigned* Arow = (const unsigned*)(A + (size_t)(mt * 16 + nl) * KP);
    const unsigned* Bbase = Wpack + ((size_t)nt0 * 32 + lane) * 8;

    float8 c[NG];
    #pragma unroll
    for (int j = 0; j < NG; ++j) c[j] = (float8){0.f,0.f,0.f,0.f,0.f,0.f,0.f,0.f};

    #pragma unroll
    for (int kt = 0; kt < KTILES; ++kt) {
        union { half16 v; uint4 q[2]; } a;
        // A fragment (16x32, k = 16h + s as in Wpack): per lane two contiguous 16B loads
        a.q[0] = *(const uint4*)(Arow + kt * 16 + 8 * h);
        a.q[1] = *(const uint4*)(Arow + kt * 16 + 8 * h + 4);
        #pragma unroll
        for (int j = 0; j < NG; ++j) {
            union { half16 v; uint4 q[2]; } b;
            const uint4* bq = (const uint4*)(Bbase + ((size_t)kt * NTILES + j) * 256);
            b.q[0] = bq[0];
            b.q[1] = bq[1];
            c[j] = __builtin_amdgcn_wmma_f32_16x16x32_f16(false, a.v, false, b.v,
                                                          (short)0, c[j], false, false);
        }
    }

    #pragma unroll
    for (int j = 0; j < NG; ++j) {
        int n = (nt0 + j) * 16 + nl;
        float bv = (n < Nvalid) ? bias[n] : 0.f;
        #pragma unroll
        for (int r = 0; r < 8; ++r) {
            int m = mt * 16 + r + 8 * h;         // D layout: VGPR r -> M = r (+8 for hi lanes)
            float val = c[j][r] + bv;
            if (act == 1) val = val > 0.f ? val : 0.f;
            if (n >= Nvalid || m >= Mvalid) val = 0.f;
            Out[(size_t)m * NP + n] = (_Float16)val;
        }
    }
}

__global__ void np_zero_f32(float* __restrict__ p, int n) {
    int t = blockIdx.x * blockDim.x + threadIdx.x;
    if (t < n) p[t] = 0.f;
}

// Segment-sum: agg[owner[e]][n] += h2[e][n]  (the "mask @ h" GEMM done sparse)
__global__ void np_scatter_add(const _Float16* __restrict__ h2, const int* __restrict__ owner,
                               float* __restrict__ agg, int Nvalid) {
    int t = blockIdx.x * blockDim.x + threadIdx.x;  // E * (KP/2)
    if (t >= N_EDGES * (KP / 2)) return;
    int e = t / (KP / 2), d = t % (KP / 2);
    int n0 = 2 * d;
    if (n0 >= Nvalid) return;
    int o = owner[e];
    float f0 = (float)h2[(size_t)e * KP + n0];
    float f1 = (float)h2[(size_t)e * KP + n0 + 1];
    atomicAdd(agg + (size_t)o * KP + n0, f0);
    atomicAdd(agg + (size_t)o * KP + n0 + 1, f1);
}

__global__ void np_f32_to_f16(const float* __restrict__ in, _Float16* __restrict__ out, int n) {
    int t = blockIdx.x * blockDim.x + threadIdx.x;
    if (t < n) out[t] = (_Float16)in[t];
}

// Fused final classifier: sigmoid(l4[m,0:100] . Wc + bc)
__global__ void np_predict(const _Float16* __restrict__ l4, const float* __restrict__ Wc,
                           const float* __restrict__ bc, float* __restrict__ out, int M) {
    int m = blockIdx.x * blockDim.x + threadIdx.x;
    if (m >= M) return;
    float acc = 0.f;
    #pragma unroll 4
    for (int k = 0; k < DEC; ++k)
        acc += (float)l4[(size_t)m * NP_L4 + k] * Wc[k];
    acc += bc[0];
    out[m] = 1.f / (1.f + expf(-acc));
}

// ---------------- launch ----------------
extern "C" void kernel_launch(void* const* d_in, const int* in_sizes, int n_in,
                              void* d_out, int out_size, void* d_ws, size_t ws_size,
                              hipStream_t stream) {
    const float* dvs   = (const float*)d_in[0];
    const float* dfs   = (const float*)d_in[1];
    const float* ef    = (const float*)d_in[2];
    const float* meta  = (const float*)d_in[3];
    const float* vmask = (const float*)d_in[4];
    const float* fmask = (const float*)d_in[5];
    const float* bvmask= (const float*)d_in[6];

    char* ws = (char*)d_ws;
    size_t off = 0;
    auto alloc = [&](size_t bytes) -> void* {
        void* p = ws + off;
        off = (off + bytes + 255) & ~(size_t)255;
        return p;
    };
    int*       idx_var = (int*)alloc(N_EDGES * 4);
    int*       idx_fun = (int*)alloc(N_EDGES * 4);
    int*       g_of_v  = (int*)alloc(1504 * 4);
    _Float16*  Av      = (_Float16*)alloc((size_t)N_EDGES * KP * 2);
    _Float16*  Af      = (_Float16*)alloc((size_t)N_EDGES * KP * 2);
    _Float16*  h1      = (_Float16*)alloc((size_t)N_EDGES * KP * 2);
    _Float16*  h2      = (_Float16*)alloc((size_t)N_EDGES * KP * 2);
    float*     agg32   = (float*)alloc((size_t)N_FUNCS * KP * 4);
    _Float16*  agg16   = (_Float16*)alloc((size_t)N_FUNCS * KP * 2);
    _Float16*  aout    = (_Float16*)alloc((size_t)N_FUNCS * KP * 2);
    _Float16*  l4      = (_Float16*)alloc((size_t)N_FUNCS * NP_L4 * 2);
    unsigned*  pW1     = (unsigned*)alloc(KTILES * 10 * 256 * 4);
    unsigned*  pW2     = (unsigned*)alloc(KTILES * 10 * 256 * 4);
    unsigned*  pWa1    = (unsigned*)alloc(KTILES * 10 * 256 * 4);
    unsigned*  pWa2    = (unsigned*)alloc(KTILES * 8 * 256 * 4);

    const int T = 256;
    auto nb = [](long long n, int t) { return (unsigned)((n + t - 1) / t); };

    // 1) recover sparse structure from the dense one-hot masks (dominant HBM traffic)
    np_extract_onehot<<<nb((long long)N_VARS * N_EDGES / 4, T), T, 0, stream>>>(vmask, idx_var, N_VARS, N_EDGES, 0);
    np_extract_onehot<<<nb((long long)N_FUNCS * N_EDGES / 4, T), T, 0, stream>>>(fmask, idx_fun, N_FUNCS, N_EDGES, 0);
    np_extract_onehot<<<nb((long long)N_VARS * 16 / 4, T), T, 0, stream>>>(bvmask, g_of_v, N_VARS, 16, 1);

    // 2) pack concatenated edge inputs to f16 (graph_feat gather fused)
    np_pack_inputs<<<nb((long long)N_EDGES * (KP / 2), T), T, 0, stream>>>(
        dvs, dfs, ef, meta, idx_var, g_of_v, Av, Af);

    // 3) two sides: v then f (buffers reused sequentially, stream-ordered)
    for (int side = 0; side < 2; ++side) {
        int wb = 7 + side * 10;
        const float* W1m = (const float*)d_in[wb + 0];
        const float* b1m = (const float*)d_in[wb + 1];
        const float* W2m = (const float*)d_in[wb + 2];
        const float* b2m = (const float*)d_in[wb + 3];
        const float* W1a = (const float*)d_in[wb + 4];
        const float* b1a = (const float*)d_in[wb + 5];
        const float* W2a = (const float*)d_in[wb + 6];
        const float* b2a = (const float*)d_in[wb + 7];
        const float* Wc  = (const float*)d_in[wb + 8];
        const float* bc  = (const float*)d_in[wb + 9];

        const _Float16* Ain = side ? Af : Av;
        const int* owner    = side ? idx_fun : idx_var;
        int Mvalid = side ? N_FUNCS : N_VARS;          // 6000 / 1500
        int Mpad   = side ? N_FUNCS : 1504;
        int Mtiles = Mpad / 16;                        // 375 / 94
        float* outp = (float*)d_out + (side ? N_VARS : 0);

        np_pack_w<<<nb(KTILES * 10 * 256, T), T, 0, stream>>>(W1m, pW1, 133, MEMH, KTILES, 10);
        np_pack_w<<<nb(KTILES * 10 * 256, T), T, 0, stream>>>(W2m, pW2, MEMH, MEMH, KTILES, 10);
        np_pack_w<<<nb(KTILES * 10 * 256, T), T, 0, stream>>>(W1a, pWa1, MEMH, MEMH, KTILES, 10);
        np_pack_w<<<nb(KTILES * 8 * 256, T), T, 0, stream>>>(W2a, pWa2, MEMH, DEC, KTILES, 8);

        // edge MLP: h1 = relu(s @ W1m + b); h2 = relu(h1 @ W2m + b)
        np_wmma_gemm<5, 10, NP_H><<<dim3(2, N_EDGES / 16), 32, 0, stream>>>(Ain, pW1, b1m, h1, N_EDGES, MEMH, 1);
        np_wmma_gemm<5, 10, NP_H><<<dim3(2, N_EDGES / 16), 32, 0, stream>>>(h1, pW2, b2m, h2, N_EDGES, MEMH, 1);

        // sparse aggregation (mask @ h)
        np_zero_f32<<<nb((long long)Mpad * KP, T), T, 0, stream>>>(agg32, Mpad * KP);
        np_scatter_add<<<nb((long long)N_EDGES * (KP / 2), T), T, 0, stream>>>(h2, owner, agg32, MEMH);
        np_f32_to_f16<<<nb((long long)Mpad * KP, T), T, 0, stream>>>(agg32, agg16, Mpad * KP);

        // node MLP: a = relu(agg @ W1a + b); l4 = a @ W2a + b
        np_wmma_gemm<5, 10, NP_H><<<dim3(2, Mtiles), 32, 0, stream>>>(agg16, pWa1, b1a, aout, Mvalid, MEMH, 1);
        np_wmma_gemm<4, 8, NP_L4><<<dim3(2, Mtiles), 32, 0, stream>>>(aout, pWa2, b2a, l4, Mvalid, DEC, 0);

        // sigmoid(l4 @ Wc + bc)
        np_predict<<<nb(Mvalid, T), T, 0, stream>>>(l4, Wc, bc, outp, Mvalid);
    }
}
